// MultiHeadAttention_17179869509
// MI455X (gfx1250) — matrix operand order repaired and checked by reading, not compiled, not run on hardware
//
#include <hip/hip_runtime.h>
#include <hip/hip_bf16.h>

// ---------------------------------------------------------------------------
// MultiHeadAttention forward for B=2, T=2048, D=1024, H=16, DH=64 on gfx1250.
// bf16 WMMA (v_wmma_f32_16x16x32_bf16); fp32 accumulate; async LDS staging.
// ---------------------------------------------------------------------------

#define BB 2
#define TT 2048
#define DD 1024
#define HH 16
#define DHD 64
#define MM (BB * TT)   // 4096 rows in the projection GEMMs

typedef __attribute__((ext_vector_type(16))) __bf16 v16bf;
typedef __attribute__((ext_vector_type(8)))  __bf16 v8bf;
typedef __attribute__((ext_vector_type(8)))  float  v8f;

static __device__ __forceinline__ v16bf make_frag(const __bf16* lo, const __bf16* hi) {
  v8bf a = *(const v8bf*)lo;
  v8bf b = *(const v8bf*)hi;
  v16bf r;
#pragma unroll
  for (int i = 0; i < 8; ++i) { r[i] = a[i]; r[i + 8] = b[i]; }
  return r;
}

static __device__ __forceinline__ v8f wmma_bf16(v16bf a, v16bf b, v8f c) {
  return __builtin_amdgcn_wmma_f32_16x16x32_bf16(false, a, false, b, (short)0, c,
                                                 false, false);
}

// Async global -> LDS copy (16B per lane), ASYNCcnt-tracked.  GV addressing:
// VDST = VGPR with LDS byte offset (low 32 bits of generic shared pointer),
// VADDR = 64-bit global address VGPR pair, SADDR = off.
static __device__ __forceinline__ void async_load_b128(const void* gptr, void* lptr) {
  unsigned lds = (unsigned)(uintptr_t)lptr;
  unsigned long long ga = (unsigned long long)(uintptr_t)gptr;
  asm volatile("global_load_async_to_lds_b128 %0, %1, off"
               :: "v"(lds), "v"(ga) : "memory");
}
static __device__ __forceinline__ void wait_async0() {
  asm volatile("s_wait_asynccnt 0x0" ::: "memory");
}

// ---------------------------------------------------------------------------
// GEMM:  Out[M,N] = A[M,K] @ W[K,N] + bias  (optionally *scale)
//   CONV_A        : A is fp32 (convert to bf16 while staging); else A is bf16
//                   and is staged with async LDS copies.
//   HEADMAJOR_OUT : write bf16 to [(b*H+h)*T + t]*DH + dh  (for Q/K/V);
//                   else write fp32 row-major [M,N] (final projection).
// Workgroup 256 threads (8 waves), 128x128 output tile, K-step 32.
// Wave w: rows (w>>1)*32..+31, cols (w&1)*64..+63  -> 2x4 accumulators.
// ---------------------------------------------------------------------------
template <bool CONV_A, bool HEADMAJOR_OUT>
__global__ __launch_bounds__(256) void gemm_wmma_kernel(
    const void* __restrict__ Av, const float* __restrict__ W,
    const float* __restrict__ bias, void* __restrict__ Outv,
    int M, int N, int K, float scale) {
  __shared__ __bf16 lA[128][32];   // A tile, row-major          (8 KB)
  __shared__ __bf16 lBt[128][32];  // W tile, transposed [n][k]  (8 KB)

  const int tid  = threadIdx.x;
  const int wv   = tid >> 5;
  const int lane = tid & 31;
  const int half = lane >> 4;
  const int ln   = lane & 15;

  const int n0 = blockIdx.x * 128;
  const int m0 = blockIdx.y * 128;
  const int rw = wv >> 1, cw = wv & 1;

  v8f acc[2][4];
#pragma unroll
  for (int mi = 0; mi < 2; ++mi)
#pragma unroll
    for (int ni = 0; ni < 4; ++ni)
#pragma unroll
      for (int r = 0; r < 8; ++r) acc[mi][ni][r] = 0.0f;

  const int arow = tid >> 1, aseg = tid & 1;          // A tile: 128 rows x 2 segs
  const int bk = tid >> 3, bseg = tid & 7;            // W tile: 32 k x 8 segs

  for (int kk = 0; kk < K; kk += 32) {
    // ---- stage A tile ----
    if (CONV_A) {
      const float* ap = (const float*)Av + (size_t)(m0 + arow) * K + kk + aseg * 16;
#pragma unroll
      for (int i = 0; i < 16; ++i) lA[arow][aseg * 16 + i] = (__bf16)ap[i];
    } else {
      const __bf16* ap = (const __bf16*)Av + (size_t)(m0 + arow) * K + kk + aseg * 16;
      async_load_b128(ap,     &lA[arow][aseg * 16]);      // bypass VGPRs
      async_load_b128(ap + 8, &lA[arow][aseg * 16 + 8]);
    }
    // ---- stage W tile transposed: lBt[n][k] = W[kk+k][n0+n] ----
    {
      const float* wp = W + (size_t)(kk + bk) * N + n0 + bseg * 16;
#pragma unroll
      for (int i = 0; i < 16; ++i) lBt[bseg * 16 + i][bk] = (__bf16)wp[i];
    }
    if (!CONV_A) wait_async0();
    __syncthreads();

    // ---- fragments + 8 WMMAs ----
    v16bf afr[2], bfr[4];
#pragma unroll
    for (int mi = 0; mi < 2; ++mi) {
      const int row = rw * 32 + mi * 16 + ln;
      afr[mi] = make_frag(&lA[row][16 * half], &lA[row][16 * half + 8]);
    }
#pragma unroll
    for (int ni = 0; ni < 4; ++ni) {
      const int n = cw * 64 + ni * 16 + ln;
      bfr[ni] = *(const v16bf*)&lBt[n][16 * half];  // 32B contiguous
    }
#pragma unroll
    for (int mi = 0; mi < 2; ++mi)
#pragma unroll
      for (int ni = 0; ni < 4; ++ni)
        acc[mi][ni] = wmma_bf16(afr[mi], bfr[ni], acc[mi][ni]);
    __syncthreads();
  }

  // ---- epilogue ----
#pragma unroll
  for (int mi = 0; mi < 2; ++mi)
#pragma unroll
    for (int ni = 0; ni < 4; ++ni) {
      const int ng = n0 + cw * 64 + ni * 16 + ln;
      const float bv = bias[ng];
#pragma unroll
      for (int r = 0; r < 8; ++r) {
        const int mg = m0 + rw * 32 + mi * 16 + r + 8 * half;
        float val = acc[mi][ni][r] + bv;
        if (HEADMAJOR_OUT) {
          val *= scale;
          const int b = mg / TT, t = mg % TT;
          const int h = ng >> 6, dh = ng & 63;
          ((__bf16*)Outv)[(((size_t)b * HH + h) * TT + t) * DHD + dh] = (__bf16)val;
        } else {
          ((float*)Outv)[(size_t)mg * N + ng] = val;
        }
      }
    }
}

// ---------------------------------------------------------------------------
// Flash attention (causal, online softmax).
// Head-major bf16 Q/K/V in, bf16 attn-out (row-major [b*T+t][h*64+dh]) out.
// grid = (B*H, T/128); 256 threads; wave w owns query rows qb*128 + w*16 ..+15.
// Key blocks of 64; K block staged with ASYNC LDS copies; V^T staged via
// VGPR transpose in parallel; next block prefetched into cache.
// Softmax scale (1/8) is pre-folded into Q.
// ---------------------------------------------------------------------------
__global__ __launch_bounds__(256) void flash_attn_kernel(
    const __bf16* __restrict__ Qh, const __bf16* __restrict__ Kh,
    const __bf16* __restrict__ Vh, __bf16* __restrict__ AO) {
  __shared__ __bf16 lK[64][64];        // K block  [key][dh]   (8 KB)
  __shared__ __bf16 lVt[64][64];       // V block T [dh][key]  (8 KB)
  __shared__ __bf16 lP[8][16][64];     // per-wave P staging   (16 KB)

  const int tid  = threadIdx.x;
  const int wv   = tid >> 5;
  const int lane = tid & 31;
  const int half = lane >> 4;
  const int ln   = lane & 15;

  const int bh = blockIdx.x;           // 0..31  (b*H + h)
  const int qb = blockIdx.y;           // 0..15
  const int t_base = qb * 128 + wv * 16;

  // ---- Q fragments (A-layout) straight from global, scaled already ----
  const __bf16* qrow = Qh + ((size_t)bh * TT + t_base + ln) * DHD;
  v16bf qf[2];
#pragma unroll
  for (int f = 0; f < 2; ++f)
    qf[f] = make_frag(qrow + 32 * f + 16 * half, qrow + 32 * f + 16 * half + 8);

  float mrun[8], lrun[8];
  v8f Oacc[4];
#pragma unroll
  for (int r = 0; r < 8; ++r) { mrun[r] = -1e30f; lrun[r] = 0.0f; }
#pragma unroll
  for (int j = 0; j < 4; ++j)
#pragma unroll
    for (int r = 0; r < 8; ++r) Oacc[j][r] = 0.0f;

  const int krow = tid >> 2, kseg = tid & 3;  // coop loaders: 64 rows x 4 segs
  const int nkb = 2 * qb + 2;

  for (int kb = 0; kb < nkb; ++kb) {
    // ---- K block: async global -> LDS (no VGPR round-trip) ----
    {
      const __bf16* kp = Kh + ((size_t)bh * TT + kb * 64 + krow) * DHD + kseg * 16;
      async_load_b128(kp,     &lK[krow][kseg * 16]);
      async_load_b128(kp + 8, &lK[krow][kseg * 16 + 8]);
    }
    // ---- V block: load + transpose into lVt while K streams in ----
    {
      const __bf16* vp = Vh + ((size_t)bh * TT + kb * 64 + krow) * DHD + kseg * 16;
      v8bf v0 = *(const v8bf*)(vp);
      v8bf v1 = *(const v8bf*)(vp + 8);
#pragma unroll
      for (int i = 0; i < 8; ++i) {
        lVt[kseg * 16 + i][krow]     = v0[i];
        lVt[kseg * 16 + 8 + i][krow] = v1[i];
      }
    }
    // ---- prefetch next key block (global_prefetch_b8) ----
    if (kb + 1 < nkb) {
      const size_t nxt = ((size_t)bh * TT + (kb + 1) * 64 + krow) * DHD + kseg * 16;
      __builtin_prefetch(Kh + nxt, 0, 1);
      __builtin_prefetch(Vh + nxt, 0, 1);
    }
    wait_async0();
    __syncthreads();

    if (kb * 64 <= t_base + 15) {  // uniform per wave: any unmasked key?
      // ---- S = Q @ K^T  (k-dim = dh, 2 steps of 32) ----
      v8f S[4];
#pragma unroll
      for (int j = 0; j < 4; ++j)
#pragma unroll
        for (int r = 0; r < 8; ++r) S[j][r] = 0.0f;
#pragma unroll
      for (int s = 0; s < 2; ++s)
#pragma unroll
        for (int j = 0; j < 4; ++j) {
          v16bf kfr = *(const v16bf*)&lK[j * 16 + ln][32 * s + 16 * half];
          S[j] = wmma_bf16(qf[s], kfr, S[j]);
        }

      // ---- causal mask (boundary block only needs it, but cheap) ----
      if (kb * 64 + 63 > t_base) {
#pragma unroll
        for (int j = 0; j < 4; ++j) {
          const int ng = kb * 64 + j * 16 + ln;
#pragma unroll
          for (int r = 0; r < 8; ++r) {
            const int mg = t_base + r + 8 * half;
            if (ng > mg) S[j][r] = -1e30f;
          }
        }
      }

      // ---- online softmax per row (16-lane half reductions) ----
#pragma unroll
      for (int r = 0; r < 8; ++r) {
        float mx = -1e30f;
#pragma unroll
        for (int j = 0; j < 4; ++j) mx = fmaxf(mx, S[j][r]);
#pragma unroll
        for (int off = 8; off >= 1; off >>= 1) mx = fmaxf(mx, __shfl_xor(mx, off, 32));
        const float mnew = fmaxf(mrun[r], mx);
        const float alpha = __expf(mrun[r] - mnew);
        float rs = 0.0f;
#pragma unroll
        for (int j = 0; j < 4; ++j) {
          const float p = __expf(S[j][r] - mnew);
          S[j][r] = p;
          rs += p;
        }
#pragma unroll
        for (int off = 8; off >= 1; off >>= 1) rs += __shfl_xor(rs, off, 32);
        lrun[r] = lrun[r] * alpha + rs;
        mrun[r] = mnew;
#pragma unroll
        for (int j = 0; j < 4; ++j) Oacc[j][r] *= alpha;
      }

      // ---- P: C-layout -> A-layout via per-wave LDS strip ----
#pragma unroll
      for (int j = 0; j < 4; ++j)
#pragma unroll
        for (int r = 0; r < 8; ++r)
          lP[wv][r + 8 * half][j * 16 + ln] = (__bf16)S[j][r];

      // ---- O += P @ V  (k-dim = key, 2 steps of 32) ----
#pragma unroll
      for (int s = 0; s < 2; ++s) {
        v16bf pf = make_frag(&lP[wv][ln][32 * s + 16 * half],
                             &lP[wv][ln][32 * s + 16 * half + 8]);
#pragma unroll
        for (int j = 0; j < 4; ++j) {
          v16bf vfr = *(const v16bf*)&lVt[j * 16 + ln][32 * s + 16 * half];
          Oacc[j] = wmma_bf16(pf, vfr, Oacc[j]);
        }
      }
    }
    __syncthreads();
  }

  // ---- normalize and store (row-major [b*T+t][h*64+dh], bf16) ----
  const int b = bh >> 4, h = bh & 15;
#pragma unroll
  for (int j = 0; j < 4; ++j)
#pragma unroll
    for (int r = 0; r < 8; ++r) {
      const int t = t_base + r + 8 * half;
      const int col = h * 64 + j * 16 + ln;
      const float val = Oacc[j][r] / lrun[r];
      AO[((size_t)b * TT + t) * DD + col] = (__bf16)val;
    }
}

// ---------------------------------------------------------------------------
extern "C" void kernel_launch(void* const* d_in, const int* in_sizes, int n_in,
                              void* d_out, int out_size, void* d_ws, size_t ws_size,
                              hipStream_t stream) {
  const float* q  = (const float*)d_in[0];
  const float* k  = (const float*)d_in[1];
  const float* v  = (const float*)d_in[2];
  const float* Wq = (const float*)d_in[3];
  const float* bq = (const float*)d_in[4];
  const float* Wk = (const float*)d_in[5];
  const float* bk = (const float*)d_in[6];
  const float* Wv = (const float*)d_in[7];
  const float* bv = (const float*)d_in[8];
  const float* Wo = (const float*)d_in[9];
  const float* bo = (const float*)d_in[10];

  const size_t nElem = (size_t)MM * DD;  // 4096*1024
  __bf16* bfQ  = (__bf16*)d_ws;
  __bf16* bfK  = bfQ + nElem;
  __bf16* bfV  = bfK + nElem;
  __bf16* bfAO = bfV + nElem;

  const dim3 gGemm(DD / 128, MM / 128);  // (8, 32)
  const dim3 blk(256);
  const float qscale = 0.125f;  // 1/sqrt(DH=64)

  gemm_wmma_kernel<true, true><<<gGemm, blk, 0, stream>>>(q, Wq, bq, bfQ, MM, DD, DD, qscale);
  gemm_wmma_kernel<true, true><<<gGemm, blk, 0, stream>>>(k, Wk, bk, bfK, MM, DD, DD, 1.0f);
  gemm_wmma_kernel<true, true><<<gGemm, blk, 0, stream>>>(v, Wv, bv, bfV, MM, DD, DD, 1.0f);

  flash_attn_kernel<<<dim3(BB * HH, TT / 128), blk, 0, stream>>>(bfQ, bfK, bfV, bfAO);

  gemm_wmma_kernel<false, false><<<gGemm, blk, 0, stream>>>(bfAO, Wo, bo, d_out, MM, DD, DD, 1.0f);
}
